// SelfAttention_81458349736252
// MI455X (gfx1250) — matrix operand order repaired and checked by reading, not compiled, not run on hardware
//
#include <hip/hip_runtime.h>

typedef __attribute__((ext_vector_type(16))) __bf16          v16bf;
typedef __attribute__((ext_vector_type(8)))  float           v8f;
typedef __attribute__((ext_vector_type(8)))  unsigned short  ush8;
typedef __attribute__((ext_vector_type(4)))  int             i4;

#define AS1 __attribute__((address_space(1)))
#define AS3 __attribute__((address_space(3)))

#define LDSS 40   // padded LDS row stride in bf16 elements (80B = 20 dwords; gcd(20,64)=4 -> conflict-free frag reads)
#define BM 256    // block tile M
#define BN 128    // block tile N  (8 waves: 4 along M x 2 along N, 64x64 per wave)

#if __has_builtin(__builtin_amdgcn_global_load_async_to_lds_b128) && \
    __has_builtin(__builtin_amdgcn_s_wait_asynccnt)
#define USE_ASYNC_LDS 1
#else
#define USE_ASYNC_LDS 0
#endif

__device__ __forceinline__ unsigned short f2bf(float f) {
  union { float f; unsigned u; } c; c.f = f;
  unsigned u = c.u;
  u += 0x7FFFu + ((u >> 16) & 1u);       // round-to-nearest-even
  return (unsigned short)(u >> 16);
}

// ---------------------------------------------------------------- convert x
__global__ __launch_bounds__(256) void cvt_f32_bf16(const float* __restrict__ src,
                                                    unsigned short* __restrict__ dst,
                                                    int n4) {
  int i = blockIdx.x * 256 + threadIdx.x;
  if (i < n4) {
    float4 v = ((const float4*)src)[i];
    struct alignas(8) US4 { unsigned short a, b, c, d; };
    US4 o{ f2bf(v.x), f2bf(v.y), f2bf(v.z), f2bf(v.w) };
    ((US4*)dst)[i] = o;
  }
}

// ---------------------------------------------------------------- transpose
// dst[c][r] = src[r][c] (bf16 out), 32x32 LDS tiles, batched over z.
template <bool F32SRC>
__global__ __launch_bounds__(256) void transpose_to_bf16(
    const void* __restrict__ src, unsigned short* __restrict__ dst,
    int R, int C, long s_bs, long d_bs) {
  __shared__ unsigned short t[32][33];
  const int  z  = blockIdx.z;
  const int  tx = threadIdx.x & 31;
  const int  ty = threadIdx.x >> 5;              // 0..7
  const long r0 = (long)blockIdx.y * 32;
  const long c0 = (long)blockIdx.x * 32;
#pragma unroll
  for (int i = 0; i < 32; i += 8) {
    const long r = r0 + ty + i;
    unsigned short hv;
    if (F32SRC) hv = f2bf(((const float*)src)[(long)z * s_bs + r * C + c0 + tx]);
    else        hv = ((const unsigned short*)src)[(long)z * s_bs + r * C + c0 + tx];
    t[ty + i][tx] = hv;
  }
  __syncthreads();
#pragma unroll
  for (int i = 0; i < 32; i += 8)
    dst[(long)z * d_bs + (c0 + ty + i) * R + r0 + tx] = t[tx][ty + i];
}

// ---------------------------------------------------------------- GEMM
// C[BM*gy, BN*gx] = A(bf16 [M][K]) * B(bf16 [N][K], K contiguous), K-step 32.
// 8 wave32s, 64x64 tile per wave: 16 ds_load_b128 feed 16 v_wmma per K-step
// (1.0 DS op / WMMA).  Tiles staged with GLOBAL_LOAD_ASYNC_TO_LDS_B128
// (ASYNCcnt) when available; double-buffered LDS (60 KB of the 320 KB WGP pool).
// OUT_BF16: (acc + bias[col]) * scale -> bf16 ; else plain f32 store.
template <bool OUT_BF16>
__global__ __launch_bounds__(256) void gemm_bf16_256x128(
    const unsigned short* __restrict__ A, long lda, long a_bs,
    const unsigned short* __restrict__ B, long ldb, long b_bs,
    void* __restrict__ C, long ldc, long c_bs,
    const float* __restrict__ bias0, const float* __restrict__ bias1,
    const float* __restrict__ bias2, float scale_z0, int Kdim) {
  __shared__ alignas(16) unsigned short lsA[2][BM * LDSS];
  __shared__ alignas(16) unsigned short lsB[2][BN * LDSS];

  const int tid  = threadIdx.x;
  const int lane = tid & 31;
  const int w    = tid >> 5;
  const int wm   = w >> 1;       // 4 waves along M (64 rows each)
  const int wn   = w & 1;        // 2 waves along N (64 cols each)
  const int z    = blockIdx.z;

  const unsigned short* Ab = A + (long)z * a_bs + (long)blockIdx.y * BM * lda;
  const unsigned short* Bb = B + (long)z * b_bs + (long)blockIdx.x * BN * ldb;

  // 16B chunks: A tile 256x32 -> 1024 chunks (4/thread); B tile 128x32 -> 512 (2/thread)
  int rA[4], kA[4], rB[2], kB[2];
#pragma unroll
  for (int p = 0; p < 4; ++p) {
    const int c = tid + 256 * p;
    rA[p] = c >> 2; kA[p] = (c & 3) << 3;
  }
#pragma unroll
  for (int p = 0; p < 2; ++p) {
    const int c = tid + 256 * p;
    rB[p] = c >> 2; kB[p] = (c & 3) << 3;
  }

#if USE_ASYNC_LDS
  auto gc = [](const unsigned short* p) { return (AS1 i4*)(uintptr_t)p; };
  auto lc = [](unsigned short* p) { return (AS3 i4*)(unsigned)(uintptr_t)p; };
  auto loadTile = [&](int buf, int k0) {
#pragma unroll
    for (int p = 0; p < 4; ++p)
      __builtin_amdgcn_global_load_async_to_lds_b128(
          gc(Ab + (long)rA[p] * lda + k0 + kA[p]),
          lc(&lsA[buf][rA[p] * LDSS + kA[p]]), 0, 0);
#pragma unroll
    for (int p = 0; p < 2; ++p)
      __builtin_amdgcn_global_load_async_to_lds_b128(
          gc(Bb + (long)rB[p] * ldb + k0 + kB[p]),
          lc(&lsB[buf][rB[p] * LDSS + kB[p]]), 0, 0);
  };
  loadTile(0, 0);
  __builtin_amdgcn_s_wait_asynccnt(0);
  __syncthreads();
#else
  uint4 ra[4], rb[2];
  auto loadTileRegs = [&](int k0) {
#pragma unroll
    for (int p = 0; p < 4; ++p) ra[p] = *(const uint4*)(Ab + (long)rA[p] * lda + k0 + kA[p]);
#pragma unroll
    for (int p = 0; p < 2; ++p) rb[p] = *(const uint4*)(Bb + (long)rB[p] * ldb + k0 + kB[p]);
  };
  auto storeTile = [&](int buf) {
#pragma unroll
    for (int p = 0; p < 4; ++p) *(uint4*)(&lsA[buf][rA[p] * LDSS + kA[p]]) = ra[p];
#pragma unroll
    for (int p = 0; p < 2; ++p) *(uint4*)(&lsB[buf][rB[p] * LDSS + kB[p]]) = rb[p];
  };
  loadTileRegs(0);
  storeTile(0);
  __syncthreads();
#endif

  const v8f vzero = {0.f, 0.f, 0.f, 0.f, 0.f, 0.f, 0.f, 0.f};
  v8f acc[4][4];
#pragma unroll
  for (int i = 0; i < 4; ++i)
#pragma unroll
    for (int j = 0; j < 4; ++j) acc[i][j] = vzero;

  const int nk   = Kdim >> 5;
  const int sel  = lane >> 4;    // K-half select per ISA A/B lane layouts
  const int l15  = lane & 15;
  const int aoff = sel << 3;     // A: lanes 0-15 -> K {0..7,16..23}; 16-31 -> {8..15,24..31}
  const int koff = sel << 3;     // B: lanes 0-15 -> K {0..7,16..23}; 16-31 -> {8..15,24..31}

  for (int kt = 0; kt < nk; ++kt) {
    const int cur = kt & 1;
#if USE_ASYNC_LDS
    if (kt + 1 < nk) loadTile(cur ^ 1, (kt + 1) << 5);
#else
    if (kt + 1 < nk) loadTileRegs((kt + 1) << 5);
#endif

    union { ush8 h[2]; v16bf v; } af[4], bf[4];
#pragma unroll
    for (int i = 0; i < 4; ++i) {
      const unsigned short* p = &lsA[cur][(wm * 64 + i * 16 + l15) * LDSS + aoff];
      af[i].h[0] = *(const ush8*)p;
      af[i].h[1] = *(const ush8*)(p + 16);
    }
#pragma unroll
    for (int j = 0; j < 4; ++j) {
      const unsigned short* p = &lsB[cur][(wn * 64 + j * 16 + l15) * LDSS + koff];
      bf[j].h[0] = *(const ush8*)p;
      bf[j].h[1] = *(const ush8*)(p + 16);
    }
#pragma unroll
    for (int i = 0; i < 4; ++i)
#pragma unroll
      for (int j = 0; j < 4; ++j)
        acc[i][j] = __builtin_amdgcn_wmma_f32_16x16x32_bf16(
            false, af[i].v, false, bf[j].v, (short)0, acc[i][j], false, false);

#if USE_ASYNC_LDS
    if (kt + 1 < nk) __builtin_amdgcn_s_wait_asynccnt(0);
#else
    if (kt + 1 < nk) storeTile(cur ^ 1);
#endif
    __syncthreads();
  }

  // epilogue (C layout: vgpr v, lane l -> M = v + 8*sel, N = l15)
  const float* bias = (z == 0) ? bias0 : (z == 1 ? bias1 : bias2);
  const float  scl  = (z == 0) ? scale_z0 : 1.0f;
  const long row0 = (long)blockIdx.y * BM + wm * 64 + sel * 8;
  const long col0 = (long)blockIdx.x * BN + wn * 64 + l15;

#pragma unroll
  for (int i = 0; i < 4; ++i) {
#pragma unroll
    for (int j = 0; j < 4; ++j) {
      const long  col = col0 + j * 16;
      const float bv  = bias ? bias[col] : 0.0f;
#pragma unroll
      for (int v = 0; v < 8; ++v) {
        const long  row = row0 + i * 16 + v;
        const float val = (acc[i][j][v] + bv) * scl;
        if (OUT_BF16)
          ((unsigned short*)C)[(long)z * c_bs + row * ldc + col] = f2bf(val);
        else
          ((float*)C)[(long)z * c_bs + row * ldc + col] = val;
      }
    }
  }
}

// ---------------------------------------------------------------- softmax
// one wave32 per 2048-wide f32 row; writes bf16 probs in place (all loads
// are consumed into the row sum before any store issues -> safe overlap).
__global__ __launch_bounds__(256) void softmax_inplace_bf16(float* __restrict__ S) {
  const int  lane = threadIdx.x & 31;
  const long row  = (long)blockIdx.x * 8 + (threadIdx.x >> 5);
  float* rp = S + row * 2048;

  float4 v[16];
  float  m = -3.0e38f;
#pragma unroll
  for (int j = 0; j < 16; ++j) {
    v[j] = *(const float4*)(rp + lane * 4 + 128 * j);
    m = fmaxf(m, fmaxf(fmaxf(v[j].x, v[j].y), fmaxf(v[j].z, v[j].w)));
  }
  for (int off = 16; off >= 1; off >>= 1) m = fmaxf(m, __shfl_xor(m, off, 32));

  float sum = 0.f;
#pragma unroll
  for (int j = 0; j < 16; ++j) {
    v[j].x = __expf(v[j].x - m);
    v[j].y = __expf(v[j].y - m);
    v[j].z = __expf(v[j].z - m);
    v[j].w = __expf(v[j].w - m);
    sum += v[j].x + v[j].y + v[j].z + v[j].w;
  }
  for (int off = 16; off >= 1; off >>= 1) sum += __shfl_xor(sum, off, 32);
  const float inv = 1.0f / sum;

  unsigned short* op = (unsigned short*)rp;
#pragma unroll
  for (int j = 0; j < 16; ++j) {
    struct alignas(8) US4 { unsigned short a, b, c, d; };
    US4 o{ f2bf(v[j].x * inv), f2bf(v[j].y * inv),
           f2bf(v[j].z * inv), f2bf(v[j].w * inv) };
    *(US4*)(op + lane * 4 + 128 * j) = o;
  }
}

// ---------------------------------------------------------------- launch
extern "C" void kernel_launch(void* const* d_in, const int* in_sizes, int n_in,
                              void* d_out, int out_size, void* d_ws, size_t ws_size,
                              hipStream_t stream) {
  (void)in_sizes; (void)n_in; (void)out_size; (void)ws_size;
  const float* x  = (const float*)d_in[0];
  const float* Wq = (const float*)d_in[1];
  const float* bq = (const float*)d_in[2];
  const float* Wk = (const float*)d_in[3];
  const float* bk = (const float*)d_in[4];
  const float* Wv = (const float*)d_in[5];
  const float* bv = (const float*)d_in[6];
  float* out = (float*)d_out;

  // workspace layout (bf16 unless noted):
  unsigned short* xb = (unsigned short*)d_ws;            // [8192,1024]          16 MB
  unsigned short* Wt = xb + (long)8192 * 1024;           // [3][n=1024][k=1024]   6 MB
  unsigned short* Qb = Wt + (long)3 * 1024 * 1024;       // [8192,1024]          16 MB
  unsigned short* Kb = Qb + (long)8192 * 1024;           // [8192,1024]          16 MB
  unsigned short* Vb = Kb + (long)8192 * 1024;           // [8192,1024]          16 MB
  unsigned short* Vt = Vb + (long)8192 * 1024;           // [4][h=1024][key=2048] 16 MB
  float*          Sb = (float*)(Vt + (long)8192 * 1024); // [4,2048,2048] f32    64 MB

  // 1) x -> bf16 ; W -> bf16 transposed to [n][k] so GEMM B reads are K-contiguous
  cvt_f32_bf16<<<8192, 256, 0, stream>>>(x, xb, 8192 * 1024 / 4);
  transpose_to_bf16<true><<<dim3(32, 32, 1), 256, 0, stream>>>(Wq, Wt,               1024, 1024, 0, 0);
  transpose_to_bf16<true><<<dim3(32, 32, 1), 256, 0, stream>>>(Wk, Wt + 1048576,     1024, 1024, 0, 0);
  transpose_to_bf16<true><<<dim3(32, 32, 1), 256, 0, stream>>>(Wv, Wt + 2 * 1048576, 1024, 1024, 0, 0);

  // 2) fused QKV projection: z selects weight/bias; Q scaled by 1/sqrt(1024)
  gemm_bf16_256x128<true><<<dim3(8, 32, 3), 256, 0, stream>>>(
      xb, 1024, 0, Wt, 1024, 1048576, Qb, 1024, 8388608,
      bq, bk, bv, 1.0f / 32.0f, 1024);

  // 3) V -> Vt [h][key] per batch so PV GEMM B reads are K(=key)-contiguous
  transpose_to_bf16<false><<<dim3(32, 64, 4), 256, 0, stream>>>(
      Vb, Vt, 2048, 1024, 2097152, 2097152);

  // 4) scores = Qs @ K^T per batch (K is [key][h]: already K(=h)-contiguous)
  gemm_bf16_256x128<false><<<dim3(16, 8, 4), 256, 0, stream>>>(
      Qb, 1024, 2097152, Kb, 1024, 2097152, Sb, 2048, 4194304,
      nullptr, nullptr, nullptr, 1.0f, 1024);

  // 5) row softmax, bf16 probs in place (pitch becomes 4096 bf16/row)
  softmax_inplace_bf16<<<1024, 256, 0, stream>>>(Sb);

  // 6) out = P @ Vt per batch
  gemm_bf16_256x128<false><<<dim3(8, 8, 4), 256, 0, stream>>>(
      (const unsigned short*)Sb, 4096, 8388608, Vt, 2048, 2097152,
      out, 1024, 2097152, nullptr, nullptr, nullptr, 1.0f, 2048);
}
